// GATLSTMSeq2Seq_46213848105970
// MI455X (gfx1250) — matrix operand order repaired and checked by reading, not compiled, not run on hardware
//
#include <hip/hip_runtime.h>
#include <math.h>

// ---------------------------------------------------------------------------
// GAT-LSTM seq2seq for MI455X (gfx1250, wave32, WMMA).
// Dense projections -> v_wmma_f32_16x16x32_f16 (f32 accumulate) with
// fragment-order pre-swizzled f16 weights (contiguous v16h B loads).
// Edge softmax/aggregation -> L2-resident gather/scatter with float atomics.
// ---------------------------------------------------------------------------

typedef __attribute__((ext_vector_type(16))) _Float16 v16h;
typedef __attribute__((ext_vector_type(8)))  float    v8f;

__device__ __forceinline__ void atomicMaxF(float* addr, float val) {
  unsigned int* ua = reinterpret_cast<unsigned int*>(addr);
  unsigned int cur = __float_as_uint(*addr);
  while (__uint_as_float(cur) < val) {
    unsigned int prev = atomicCAS(ua, cur, __float_as_uint(val));
    if (prev == cur) break;
    cur = prev;
  }
}

__global__ void fill_kernel(float* __restrict__ p, float v, int n) {
  int i = blockIdx.x * blockDim.x + threadIdx.x;
  if (i < n) p[i] = v;
}

// Swizzle W[g][k][col] (f32, K x 64 per gate) into WMMA B-fragment order (f16),
// zero-padded to NKC*32 in K. Layout:
//   Wf[((((g*NKC + kc)*4 + ct)*2 + hsel)*16 + n)*16 + j]  = W[g][kc*32+(j&8)*2+hsel*8+(j&7)][ct*16+n]
// so in the GEMM each lane reads its whole 32-element-K B fragment as one
// contiguous 32-byte v16h.
__global__ void wswizzle_kernel(const float* __restrict__ W,
                                _Float16* __restrict__ Wf,
                                int G, int K, int NKC) {
  int i = blockIdx.x * blockDim.x + threadIdx.x;
  int total = G * NKC * 4 * 2 * 16 * 16;
  if (i >= total) return;
  int j = i & 15;      int t = i >> 4;
  int n = t & 15;      t >>= 4;
  int hsel = t & 1;    t >>= 1;
  int ct = t & 3;      t >>= 2;
  int kc = t % NKC;
  int g  = t / NKC;
  int k   = kc * 32 + (j & 8) * 2 + hsel * 8 + (j & 7);
  int col = ct * 16 + n;
  float v = (k < K) ? W[((size_t)g * K + k) * 64 + col] : 0.0f;
  Wf[i] = (_Float16)v;
}

// out[g][n][64] = A[n][K] @ W[g][K][64], one wave per 16x16 tile.
// NKC = number of K=32 chunks; TAIL16: A has only 16 valid K (upper half of
// the single chunk is zero at compile time -- no predication, no divergence).
// A fragment (16-bit A 16x32 layout): lanes 0-15 row M, K = {0..7,16..23};
// lanes 16-31 row M, K = {8..15,24..31}. C/D: VGPR r -> row (hsel? r+8 : r).
template <int NKC, bool TAIL16>
__global__ void wmma_gemm_kernel(const float* __restrict__ A,
                                 const _Float16* __restrict__ Wf,
                                 float* __restrict__ out,
                                 int NB, int K, int G, int rowTiles) {
  int tile = blockIdx.x * (blockDim.x >> 5) + (threadIdx.x >> 5);
  int total = rowTiles * 4 * G;
  if (tile >= total) return;
  int g  = tile % G;
  int t2 = tile / G;
  int ct = t2 & 3;
  int rt = t2 >> 2;
  int lane = threadIdx.x & 31;
  int hsel = lane >> 4;
  int l15  = lane & 15;
  const float* arow = A + (size_t)(rt * 16 + l15) * K;
  int kbase = hsel * 8;
  v8f acc = {};
#pragma unroll
  for (int kc = 0; kc < NKC; ++kc) {
    int k0 = kc * 32;
    v16h a;
    // lower 8: K = k0 + kbase + 0..7  (contiguous, 32B aligned)
    const float4* plo = (const float4*)(arow + k0 + kbase);
    float4 lo0 = plo[0], lo1 = plo[1];
    float alo[8] = {lo0.x, lo0.y, lo0.z, lo0.w, lo1.x, lo1.y, lo1.z, lo1.w};
#pragma unroll
    for (int j = 0; j < 8; ++j) a[j] = (_Float16)alo[j];
    if (!TAIL16) {
      // upper 8: K = k0 + 16 + kbase + 0..7
      const float4* phi = (const float4*)(arow + k0 + 16 + kbase);
      float4 hi0 = phi[0], hi1 = phi[1];
      float ahi[8] = {hi0.x, hi0.y, hi0.z, hi0.w, hi1.x, hi1.y, hi1.z, hi1.w};
#pragma unroll
      for (int j = 0; j < 8; ++j) a[8 + j] = (_Float16)ahi[j];
    } else {
#pragma unroll
      for (int j = 0; j < 8; ++j) a[8 + j] = (_Float16)0.0f;
    }
    // B fragment: one contiguous 32-byte load from the swizzled weights
    const v16h* wp = (const v16h*)(Wf +
        ((((size_t)((g * NKC + kc) * 4 + ct) * 2 + hsel) * 16 + l15) * 16));
    v16h b = *wp;
    acc = __builtin_amdgcn_wmma_f32_16x16x32_f16(false, a, false, b,
                                                 (short)0, acc, false, false);
  }
  float* og = out + (size_t)g * NB * 64;
  int col = ct * 16 + l15;
#pragma unroll
  for (int r = 0; r < 8; ++r)
    og[(size_t)(rt * 16 + hsel * 8 + r) * 64 + col] = acc[r];
}

// Pass A: per-edge logits (4 gates x 2 heads) + atomic segment-max over dst.
// Edge-attr projection (8 x 64 per gate) is recomputed on the fly: cheaper
// than materializing a 131 MB ea_proj tensor, and We stays hot in cache.
__global__ void edge_logit_kernel(const int* __restrict__ src,
                                  const int* __restrict__ dst,
                                  const float* __restrict__ ea,    // E x 8
                                  const float* __restrict__ xl,    // 4 x NB x 64
                                  const float* __restrict__ xr,    // 4 x NB x 64
                                  const float* __restrict__ We,    // 4 x 8 x 64
                                  const float* __restrict__ att,   // 4 x 2 x 32
                                  float* __restrict__ logit,       // E x 8
                                  float* __restrict__ mx,          // NB x 8
                                  int E, int NB) {
  int e = blockIdx.x * (blockDim.x >> 5) + (threadIdx.x >> 5);
  if (e >= E) return;
  int lane = threadIdx.x & 31;
  int s = src[e], d = dst[e];
  float eav[8];
#pragma unroll
  for (int f = 0; f < 8; ++f) eav[f] = ea[(size_t)e * 8 + f];
#pragma unroll
  for (int g = 0; g < 4; ++g) {
    float p0 = 0.f, p1 = 0.f;
#pragma unroll
    for (int h = 0; h < 2; ++h) {
      int dch = h * 32 + lane;
      float m = xl[((size_t)g * NB + s) * 64 + dch] +
                xr[((size_t)g * NB + d) * 64 + dch];
      float ep = 0.f;
#pragma unroll
      for (int f = 0; f < 8; ++f) ep += eav[f] * We[((size_t)g * 8 + f) * 64 + dch];
      m += ep;
      m = (m > 0.f) ? m : 0.2f * m;  // leaky_relu(0.2)
      float pr = m * att[((size_t)g * 2 + h) * 32 + lane];
      if (h == 0) p0 = pr; else p1 = pr;
    }
#pragma unroll
    for (int off = 16; off > 0; off >>= 1) {
      p0 += __shfl_xor(p0, off, 32);
      p1 += __shfl_xor(p1, off, 32);
    }
    if (lane < 2) {
      float lg = (lane == 0) ? p0 : p1;
      logit[(size_t)e * 8 + g * 2 + lane] = lg;
      atomicMaxF(&mx[(size_t)d * 8 + g * 2 + lane], lg);
    }
  }
}

// Pass B: w = exp(logit - mx[dst]); den[dst] += w
__global__ void edge_softmax_kernel(const int* __restrict__ dst,
                                    const float* __restrict__ logit,
                                    const float* __restrict__ mx,
                                    float* __restrict__ w,
                                    float* __restrict__ den,
                                    int E) {
  int i = blockIdx.x * blockDim.x + threadIdx.x;
  if (i >= E * 8) return;
  int gh = i & 7;
  int e  = i >> 3;
  int d  = dst[e];
  float ww = __expf(logit[i] - mx[(size_t)d * 8 + gh]);
  w[i] = ww;
  atomicAdd(&den[(size_t)d * 8 + gh], ww);
}

// Pass C: out[g][dst] += xl[g][src] * alpha(head)
__global__ void edge_scatter_kernel(const int* __restrict__ src,
                                    const int* __restrict__ dst,
                                    const float* __restrict__ xl,
                                    const float* __restrict__ w,
                                    const float* __restrict__ den,
                                    float* __restrict__ out,
                                    int E, int NB) {
  int e = blockIdx.x * (blockDim.x >> 5) + (threadIdx.x >> 5);
  if (e >= E) return;
  int lane = threadIdx.x & 31;
  int s = src[e], d = dst[e];
#pragma unroll
  for (int g = 0; g < 4; ++g) {
    float a0 = w[(size_t)e * 8 + g * 2 + 0] / den[(size_t)d * 8 + g * 2 + 0];
    float a1 = w[(size_t)e * 8 + g * 2 + 1] / den[(size_t)d * 8 + g * 2 + 1];
    const float* xs_ = &xl[((size_t)g * NB + s) * 64];
    float*       od  = &out[((size_t)g * NB + d) * 64];
    atomicAdd(&od[lane],      xs_[lane]      * a0);
    atomicAdd(&od[lane + 32], xs_[lane + 32] * a1);
  }
}

// Fused LSTM gates + LayerNorm(Hd=64). One wave per node, 2 ch per lane.
__global__ void cell_kernel(const float* __restrict__ gx,
                            const float* __restrict__ gh,
                            const float* __restrict__ bx,
                            const float* __restrict__ bh,
                            float* __restrict__ h, float* __restrict__ c,
                            const float* __restrict__ ln_g,
                            const float* __restrict__ ln_b, int NB) {
  int node = blockIdx.x * (blockDim.x >> 5) + (threadIdx.x >> 5);
  if (node >= NB) return;
  int lane = threadIdx.x & 31;
  float hv[2];
#pragma unroll
  for (int p = 0; p < 2; ++p) {
    int dch = lane + p * 32;
    float gv[4];
#pragma unroll
    for (int g = 0; g < 4; ++g)
      gv[g] = gx[((size_t)g * NB + node) * 64 + dch] +
              gh[((size_t)g * NB + node) * 64 + dch] +
              bx[g * 64 + dch] + bh[g * 64 + dch];
    float ig = 1.f / (1.f + __expf(-gv[0]));
    float fg = 1.f / (1.f + __expf(-gv[1]));
    float og = 1.f / (1.f + __expf(-gv[2]));
    float gg = tanhf(gv[3]);
    float cc = fg * c[(size_t)node * 64 + dch] + ig * gg;
    c[(size_t)node * 64 + dch] = cc;
    hv[p] = og * tanhf(cc);
  }
  float ssum = hv[0] + hv[1];
#pragma unroll
  for (int off = 16; off > 0; off >>= 1) ssum += __shfl_xor(ssum, off, 32);
  float mu = ssum * (1.f / 64.f);
  float d0 = hv[0] - mu, d1 = hv[1] - mu;
  float vs = d0 * d0 + d1 * d1;
#pragma unroll
  for (int off = 16; off > 0; off >>= 1) vs += __shfl_xor(vs, off, 32);
  float rstd = rsqrtf(vs * (1.f / 64.f) + 1e-5f);
#pragma unroll
  for (int p = 0; p < 2; ++p) {
    int dch = lane + p * 32;
    h[(size_t)node * 64 + dch] = (hv[p] - mu) * rstd * ln_g[dch] + ln_b[dch];
  }
}

__global__ void init_prev_kernel(float* __restrict__ prev,
                                 const float* __restrict__ x_base, int NB, int F) {
  int n = blockIdx.x * blockDim.x + threadIdx.x;
  if (n < NB) prev[n] = x_base[(size_t)n * F];
}

__global__ void build_xdec_kernel(float* __restrict__ xdec,
                                  const float* __restrict__ x_base,
                                  const float* __restrict__ prev, int NB, int F) {
  int i = blockIdx.x * blockDim.x + threadIdx.x;
  if (i >= NB * F) return;
  int n = i / F, f = i - n * F;
  xdec[i] = (f == 0) ? prev[n] : x_base[i];
}

__global__ void gelu_bias_kernel(float* __restrict__ z,
                                 const float* __restrict__ b, int NB) {
  int i = blockIdx.x * blockDim.x + threadIdx.x;
  if (i >= NB * 64) return;
  float x = z[i] + b[i & 63];
  z[i] = 0.5f * x * (1.f + erff(x * 0.70710678118654752f));  // exact gelu
}

__global__ void out_col_kernel(const float* __restrict__ z,
                               const float* __restrict__ w2,
                               const float* __restrict__ b2,
                               float* __restrict__ prev,
                               float* __restrict__ outp,
                               int NB, int S, int scol) {
  int n = blockIdx.x * blockDim.x + threadIdx.x;
  if (n >= NB) return;
  float acc = b2[scol];
  const float* zr = &z[(size_t)n * 64];
#pragma unroll
  for (int k = 0; k < 64; ++k) acc += zr[k] * w2[(size_t)k * S + scol];
  prev[n] = acc;
  outp[(size_t)n * S + scol] = acc;
}

// ---------------------------------------------------------------------------

extern "C" void kernel_launch(void* const* d_in, const int* in_sizes, int n_in,
                              void* d_out, int out_size, void* d_ws, size_t ws_size,
                              hipStream_t stream) {
  (void)n_in; (void)out_size; (void)ws_size;
  const float* x_seq  = (const float*)d_in[0];
  const int*   eidx   = (const int*)  d_in[1];
  const float* ea     = (const float*)d_in[2];
  const float* Wx_l   = (const float*)d_in[3];
  const float* Wx_r   = (const float*)d_in[4];
  const float* Wx_e   = (const float*)d_in[5];
  const float* att_x  = (const float*)d_in[6];
  const float* bx     = (const float*)d_in[7];
  const float* Wh_l   = (const float*)d_in[8];
  const float* Wh_r   = (const float*)d_in[9];
  const float* Wh_e   = (const float*)d_in[10];
  const float* att_h  = (const float*)d_in[11];
  const float* bh     = (const float*)d_in[12];
  const float* ln_g   = (const float*)d_in[13];
  const float* ln_b   = (const float*)d_in[14];
  const float* out_w1 = (const float*)d_in[15];
  const float* out_b1 = (const float*)d_in[16];
  const float* out_w2 = (const float*)d_in[17];
  const float* out_b2 = (const float*)d_in[18];
  float* outp = (float*)d_out;

  const int NB = 8000, F = 16, T = 3, S = 3;
  const int E = in_sizes[1] / 2;
  const int* src = eidx;
  const int* dst = eidx + E;
  const int rowTiles = NB / 16;

  // ---- workspace carve (bump allocator, 256B aligned) ----
  char* wp = (char*)d_ws;
  auto carveB = [&](size_t bytes) -> char* {
    char* p = wp;
    wp += (bytes + 255) & ~(size_t)255;
    return p;
  };
  auto carve  = [&](size_t nf) -> float*     { return (float*)carveB(nf * 4); };
  auto carveH = [&](size_t nh) -> _Float16*  { return (_Float16*)carveB(nh * 2); };

  float* xlx   = carve((size_t)4 * NB * 64);
  float* xrx   = carve((size_t)4 * NB * 64);
  float* xlh   = carve((size_t)4 * NB * 64);
  float* xrh   = carve((size_t)4 * NB * 64);
  float* gxb   = carve((size_t)4 * NB * 64);
  float* ghb   = carve((size_t)4 * NB * 64);
  float* logit = carve((size_t)E * 8);
  float* wbuf  = carve((size_t)E * 8);
  float* mx    = carve((size_t)NB * 8);
  float* den   = carve((size_t)NB * 8);
  float* hbuf  = carve((size_t)NB * 64);
  float* cbuf  = carve((size_t)NB * 64);
  float* zbuf  = carve((size_t)NB * 64);
  float* xdec  = carve((size_t)NB * F);
  float* prev  = carve((size_t)NB);
  // swizzled f16 weight fragments: G * NKC * 4(ct) * 2(hsel) * 16 * 16 halves
  _Float16* WfxL = carveH((size_t)4 * 1 * 2048);
  _Float16* WfxR = carveH((size_t)4 * 1 * 2048);
  _Float16* WfhL = carveH((size_t)4 * 2 * 2048);
  _Float16* WfhR = carveH((size_t)4 * 2 * 2048);
  _Float16* Wf1  = carveH((size_t)1 * 2 * 2048);

  auto fill = [&](float* p, float v, int n) {
    fill_kernel<<<(n + 255) / 256, 256, 0, stream>>>(p, v, n);
  };
  auto wswz = [&](const float* W, _Float16* Wf, int G, int K, int NKC) {
    int total = G * NKC * 2048;
    wswizzle_kernel<<<(total + 255) / 256, 256, 0, stream>>>(W, Wf, G, K, NKC);
  };
  auto gemm16 = [&](const float* A, const _Float16* Wf, float* out, int G) {
    int total = rowTiles * 4 * G;
    wmma_gemm_kernel<1, true><<<(total + 7) / 8, 256, 0, stream>>>(
        A, Wf, out, NB, 16, G, rowTiles);
  };
  auto gemm64 = [&](const float* A, const _Float16* Wf, float* out, int G) {
    int total = rowTiles * 4 * G;
    wmma_gemm_kernel<2, false><<<(total + 7) / 8, 256, 0, stream>>>(
        A, Wf, out, NB, 64, G, rowTiles);
  };
  int edgeBlocks = (E + 7) / 8;  // 8 waves (edges) per block
  auto gat = [&](const float* xl, const float* xr, const float* We,
                 const float* att, float* out) {
    fill(mx, -3.4e38f, NB * 8);
    fill(den, 0.f, NB * 8);
    fill(out, 0.f, 4 * NB * 64);
    edge_logit_kernel<<<edgeBlocks, 256, 0, stream>>>(src, dst, ea, xl, xr, We,
                                                      att, logit, mx, E, NB);
    edge_softmax_kernel<<<(E * 8 + 255) / 256, 256, 0, stream>>>(dst, logit, mx,
                                                                 wbuf, den, E);
    edge_scatter_kernel<<<edgeBlocks, 256, 0, stream>>>(src, dst, xl, wbuf, den,
                                                        out, E, NB);
  };
  auto cell = [&](const float* feats_x) {
    gemm16(feats_x, WfxL, xlx, 4);
    gemm16(feats_x, WfxR, xrx, 4);
    gemm64(hbuf,    WfhL, xlh, 4);
    gemm64(hbuf,    WfhR, xrh, 4);
    gat(xlx, xrx, Wx_e, att_x, gxb);
    gat(xlh, xrh, Wh_e, att_h, ghb);
    cell_kernel<<<(NB + 7) / 8, 256, 0, stream>>>(gxb, ghb, bx, bh, hbuf, cbuf,
                                                  ln_g, ln_b, NB);
  };

  // ---- weight swizzle (once per launch; ws is not persistent) ----
  wswz(Wx_l,   WfxL, 4, 16, 1);
  wswz(Wx_r,   WfxR, 4, 16, 1);
  wswz(Wh_l,   WfhL, 4, 64, 2);
  wswz(Wh_r,   WfhR, 4, 64, 2);
  wswz(out_w1, Wf1,  1, 64, 2);

  // ---- encoder ----
  fill(hbuf, 0.f, NB * 64);
  fill(cbuf, 0.f, NB * 64);
  for (int t = 0; t < T; ++t)
    cell(x_seq + (size_t)t * NB * F);

  // ---- decoder ----
  const float* x_base = x_seq + (size_t)(T - 1) * NB * F;
  init_prev_kernel<<<(NB + 255) / 256, 256, 0, stream>>>(prev, x_base, NB, F);
  for (int s = 0; s < S; ++s) {
    build_xdec_kernel<<<(NB * F + 255) / 256, 256, 0, stream>>>(xdec, x_base, prev, NB, F);
    cell(xdec);
    gemm64(hbuf, Wf1, zbuf, 1);
    gelu_bias_kernel<<<(NB * 64 + 255) / 256, 256, 0, stream>>>(zbuf, out_b1, NB);
    out_col_kernel<<<(NB + 255) / 256, 256, 0, stream>>>(zbuf, out_w2, out_b2,
                                                         prev, outp, NB, S, s);
  }
}
